// NaryDecoderCell_47399259079247
// MI455X (gfx1250) — compile-verified, matrix operand order checked by reading, not run on hardware
//
#include <hip/hip_runtime.h>

// ---------------------------------------------------------------------------
// NaryDecoderCell fused forward for gfx1250 (MI455X), bf16 WMMA + fp32 accum.
// ---------------------------------------------------------------------------

#define DEV static __device__ __forceinline__

typedef __bf16 bf16_t;
typedef __attribute__((ext_vector_type(16))) __bf16 v16bf;
typedef __attribute__((ext_vector_type(8)))  float  v8f;

constexpr int B  = 2048;
constexpr int C  = 256;
constexpr int HS = 512;
constexpr int H2 = 1024;
constexpr int G3 = 3 * H2;   // 3072
constexpr int D  = 8;

union FragU { v16bf v; uint4 u[2]; };

// Load a 16-half (32B) WMMA fragment slice for this lane: two contiguous
// 16-byte chunks at k and k+16 (matches the CDNA5 bf16 A/B VGPR layout).
DEV v16bf load_frag16(const bf16_t* p) {
    FragU f;
    f.u[0] = *(const uint4*)(p);
    f.u[1] = *(const uint4*)(p + 16);
    return f.v;
}

DEV v8f wmma_bf16(v16bf a, v16bf b, v8f c) {
    return __builtin_amdgcn_wmma_f32_16x16x32_bf16(
        /*neg_a=*/false, a, /*neg_b=*/false, b,
        /*c_mod=*/(short)0, c, /*reuse_a=*/false, /*reuse_b=*/false);
}

DEV float sigmoid_f(float x) { return 1.0f / (1.0f + __expf(-x)); }

// ---------------------------------------------------------------------------
// Prep kernels
// ---------------------------------------------------------------------------
__global__ void cvt_bf16(const float* __restrict__ in, bf16_t* __restrict__ out, int n) {
    int i = blockIdx.x * blockDim.x + threadIdx.x;
    int stride = gridDim.x * blockDim.x;
    for (; i < n; i += stride) out[i] = (bf16_t)in[i];
}

__global__ void build_h0(const float* __restrict__ ph, const float* __restrict__ enc,
                         bf16_t* __restrict__ out) {
    int i = blockIdx.x * blockDim.x + threadIdx.x;   // B*H2 threads
    if (i >= B * H2) return;
    int b = i >> 10, col = i & (H2 - 1);
    float v = (col < HS) ? ph[b * HS + col] : enc[b * HS + col - HS];
    out[i] = (bf16_t)v;
}

// ---------------------------------------------------------------------------
// Kernel 1: fused GRU gates for all D slots.
//   For block tile (128 rows x 64 h-cols) at slot d, each wave owns 32x32.
//   acc_r = x*Wih_r + h0*Whh_r   (i_r+h_r merged in-register)
//   acc_z = merged likewise; i_n, h_n kept separate for r*h_n.
//   Epilogue: hn = (1-z)*tanh(i_n + r*h_n) + z*h0  -> bf16 workspace,
//   plus shuffle-reduced partial dot with W_p -> atomicAdd probs accumulator.
// ---------------------------------------------------------------------------
__global__ __launch_bounds__(256)
void gru_gates(const bf16_t* __restrict__ Xbf,  const bf16_t* __restrict__ H0bf,
               const bf16_t* __restrict__ Wihb, const bf16_t* __restrict__ Whhb,
               const float* __restrict__ b_ih,  const float* __restrict__ b_hh,
               const float* __restrict__ parent_h, const float* __restrict__ encoding,
               const float* __restrict__ W_p,
               bf16_t* __restrict__ hnbf, float* __restrict__ probs_acc)
{
    const int tid  = threadIdx.x;
    const int wave = tid >> 5;
    const int lane = tid & 31;
    const int l16  = lane & 15;
    const int hi   = lane >> 4;          // 0: K 0..7/16..23, rows M 0..7 ; 1: +8
    const int koff = hi ? 8 : 0;
    const int d    = blockIdx.z;
    const int m0   = blockIdx.x * 128 + (wave & 3) * 32;
    const int n0   = blockIdx.y * 64  + (wave >> 2) * 32;

    const bf16_t* Wih = Wihb + (size_t)d * G3 * C;
    const bf16_t* Whh = Whhb + (size_t)d * G3 * H2;

    v8f accR[2][2] = {}, accZ[2][2] = {}, accIN[2][2] = {}, accHN[2][2] = {};

    // ---- input-side GEMM: K = C = 256 over x / W_ih ----
    for (int k0 = 0; k0 < C; k0 += 32) {
        v16bf a[2];
#pragma unroll
        for (int mi = 0; mi < 2; ++mi)
            a[mi] = load_frag16(Xbf + (size_t)(m0 + mi * 16 + l16) * C + k0 + koff);
#pragma unroll
        for (int g = 0; g < 3; ++g) {
#pragma unroll
            for (int ni = 0; ni < 2; ++ni) {
                v16bf bfr = load_frag16(
                    Wih + (size_t)(g * H2 + n0 + ni * 16 + l16) * C + k0 + koff);
#pragma unroll
                for (int mi = 0; mi < 2; ++mi) {
                    v8f& acc = (g == 0) ? accR[mi][ni]
                             : (g == 1) ? accZ[mi][ni] : accIN[mi][ni];
                    acc = wmma_bf16(a[mi], bfr, acc);
                }
            }
        }
    }

    // ---- hidden-side GEMM: K = H2 = 1024 over h0 / W_hh ----
    for (int k0 = 0; k0 < H2; k0 += 32) {
        v16bf a[2];
#pragma unroll
        for (int mi = 0; mi < 2; ++mi)
            a[mi] = load_frag16(H0bf + (size_t)(m0 + mi * 16 + l16) * H2 + k0 + koff);
#pragma unroll
        for (int g = 0; g < 3; ++g) {
#pragma unroll
            for (int ni = 0; ni < 2; ++ni) {
                v16bf bfr = load_frag16(
                    Whh + (size_t)(g * H2 + n0 + ni * 16 + l16) * H2 + k0 + koff);
#pragma unroll
                for (int mi = 0; mi < 2; ++mi) {
                    v8f& acc = (g == 0) ? accR[mi][ni]
                             : (g == 1) ? accZ[mi][ni] : accHN[mi][ni];
                    acc = wmma_bf16(a[mi], bfr, acc);
                }
            }
        }
    }

    // ---- GRU nonlinearity epilogue ----
#pragma unroll
    for (int ni = 0; ni < 2; ++ni) {
        const int ncol = n0 + ni * 16 + l16;
        const float bir = b_ih[d * G3 + ncol],           bhr = b_hh[d * G3 + ncol];
        const float biz = b_ih[d * G3 + H2 + ncol],      bhz = b_hh[d * G3 + H2 + ncol];
        const float bin = b_ih[d * G3 + 2 * H2 + ncol],  bhn = b_hh[d * G3 + 2 * H2 + ncol];
        const float wp  = W_p[d * H2 + ncol];
        const float* h0col = (ncol < HS) ? (parent_h + ncol) : (encoding + ncol - HS);
#pragma unroll
        for (int mi = 0; mi < 2; ++mi) {
#pragma unroll
            for (int j = 0; j < 8; ++j) {
                const int brow = m0 + mi * 16 + hi * 8 + j;   // C/D layout row
                const float r  = sigmoid_f(accR[mi][ni][j] + bir + bhr);
                const float z  = sigmoid_f(accZ[mi][ni][j] + biz + bhz);
                const float nn = tanhf(accIN[mi][ni][j] + bin +
                                       r * (accHN[mi][ni][j] + bhn));
                const float h0v = h0col[(size_t)brow * HS];
                const float hv  = (1.0f - z) * nn + z * h0v;
                hnbf[((size_t)brow * D + d) * H2 + ncol] = (bf16_t)hv;

                // partial probs dot: reduce over the 16 lanes sharing this row
                float pc = hv * wp;
                pc += __shfl_xor(pc, 1, 32);
                pc += __shfl_xor(pc, 2, 32);
                pc += __shfl_xor(pc, 4, 32);
                pc += __shfl_xor(pc, 8, 32);
                if (l16 == 0) atomicAdd(&probs_acc[brow * D + d], pc);
            }
        }
    }
}

// ---------------------------------------------------------------------------
// Kernel 2: h = tanh( sum_d (hn[:,d,:] @ W_h[d]^T) + sum_d b_h[d] )
//   Single GEMM: M=2048, N=512, K=D*H2=8192 (sum over d folded into K,
//   since hn is [B, D*H2] contiguous).
// ---------------------------------------------------------------------------
__global__ __launch_bounds__(256)
void out_gemm(const bf16_t* __restrict__ hnbf, const bf16_t* __restrict__ Whb,
              const float* __restrict__ b_h, float* __restrict__ hout)
{
    const int tid  = threadIdx.x;
    const int wave = tid >> 5;
    const int lane = tid & 31;
    const int l16  = lane & 15;
    const int hi   = lane >> 4;
    const int koff = hi ? 8 : 0;
    const int m0   = blockIdx.x * 128 + (wave & 3) * 32;
    const int n0   = blockIdx.y * 64  + (wave >> 2) * 32;

    v8f acc[2][2] = {};

    for (int kk = 0; kk < D * H2; kk += 32) {
        const int dslot = kk >> 10;
        const int hh    = (kk & (H2 - 1)) + koff;
        v16bf a[2], bfr[2];
#pragma unroll
        for (int mi = 0; mi < 2; ++mi)
            a[mi] = load_frag16(hnbf + (size_t)(m0 + mi * 16 + l16) * (D * H2) + kk + koff);
#pragma unroll
        for (int ni = 0; ni < 2; ++ni)
            bfr[ni] = load_frag16(
                Whb + ((size_t)dslot * HS + n0 + ni * 16 + l16) * H2 + hh);
#pragma unroll
        for (int mi = 0; mi < 2; ++mi)
#pragma unroll
            for (int ni = 0; ni < 2; ++ni)
                acc[mi][ni] = wmma_bf16(a[mi], bfr[ni], acc[mi][ni]);
    }

#pragma unroll
    for (int ni = 0; ni < 2; ++ni) {
        const int ncol = n0 + ni * 16 + l16;
        float bias = 0.0f;
#pragma unroll
        for (int dd = 0; dd < D; ++dd) bias += b_h[dd * HS + ncol];
#pragma unroll
        for (int mi = 0; mi < 2; ++mi)
#pragma unroll
            for (int j = 0; j < 8; ++j) {
                const int brow = m0 + mi * 16 + hi * 8 + j;
                hout[(size_t)brow * HS + ncol] = tanhf(acc[mi][ni][j] + bias);
            }
    }
}

// ---------------------------------------------------------------------------
// Kernel 3: probs = sigmoid(acc + b_p[d])
// ---------------------------------------------------------------------------
__global__ void probs_fin(const float* __restrict__ acc, const float* __restrict__ b_p,
                          float* __restrict__ out) {
    int i = blockIdx.x * blockDim.x + threadIdx.x;
    if (i < B * D) out[i] = sigmoid_f(acc[i] + b_p[i & (D - 1)]);
}

// ---------------------------------------------------------------------------
// Host launch
// ---------------------------------------------------------------------------
extern "C" void kernel_launch(void* const* d_in, const int* in_sizes, int n_in,
                              void* d_out, int out_size, void* d_ws, size_t ws_size,
                              hipStream_t stream) {
    (void)in_sizes; (void)n_in; (void)out_size; (void)ws_size;

    const float* x        = (const float*)d_in[0];   // [B, C]
    const float* parent_h = (const float*)d_in[1];   // [B, HS]
    const float* encoding = (const float*)d_in[2];   // [B, HS]
    const float* W_ih     = (const float*)d_in[3];   // [D, 3*H2, C]
    const float* W_hh     = (const float*)d_in[4];   // [D, 3*H2, H2]
    const float* b_ih     = (const float*)d_in[5];   // [D, 3*H2]
    const float* b_hh     = (const float*)d_in[6];   // [D, 3*H2]
    const float* W_p      = (const float*)d_in[7];   // [D, H2]
    const float* b_p      = (const float*)d_in[8];   // [D]
    const float* W_h      = (const float*)d_in[9];   // [D, HS, H2]
    const float* b_h      = (const float*)d_in[10];  // [D, HS]

    float* outH = (float*)d_out;                     // [B, HS]
    float* outP = outH + (size_t)B * HS;             // [B, D]

    // workspace carve-out (256B aligned slabs)
    char*  w = (char*)d_ws;
    size_t o = 0;
    auto alloc = [&](size_t bytes) -> void* {
        void* p = w + o;
        o += (bytes + 255) & ~(size_t)255;
        return p;
    };
    bf16_t* Xbf  = (bf16_t*)alloc((size_t)B * C * 2);
    bf16_t* H0bf = (bf16_t*)alloc((size_t)B * H2 * 2);
    bf16_t* Wihb = (bf16_t*)alloc((size_t)D * G3 * C * 2);
    bf16_t* Whhb = (bf16_t*)alloc((size_t)D * G3 * H2 * 2);
    bf16_t* Whb  = (bf16_t*)alloc((size_t)D * HS * H2 * 2);
    bf16_t* hnbf = (bf16_t*)alloc((size_t)B * D * H2 * 2);
    float*  pacc = (float*)alloc((size_t)B * D * 4);

    hipMemsetAsync(pacc, 0, (size_t)B * D * 4, stream);

    cvt_bf16<<<2048, 256, 0, stream>>>(x,    Xbf,  B * C);
    cvt_bf16<<<8192, 256, 0, stream>>>(W_ih, Wihb, D * G3 * C);
    cvt_bf16<<<8192, 256, 0, stream>>>(W_hh, Whhb, D * G3 * H2);
    cvt_bf16<<<8192, 256, 0, stream>>>(W_h,  Whb,  D * HS * H2);
    build_h0<<<(B * H2) / 256, 256, 0, stream>>>(parent_h, encoding, H0bf);

    gru_gates<<<dim3(B / 128, H2 / 64, D), 256, 0, stream>>>(
        Xbf, H0bf, Wihb, Whhb, b_ih, b_hh, parent_h, encoding, W_p, hnbf, pacc);

    out_gemm<<<dim3(B / 128, HS / 64), 256, 0, stream>>>(hnbf, Whb, b_h, outH);

    probs_fin<<<(B * D) / 256, 256, 0, stream>>>(pacc, b_p, outP);
}
